// BSplineKANLayer_42623255445537
// MI455X (gfx1250) — matrix operand order repaired and checked by reading, not compiled, not run on hardware
//
#include <hip/hip_runtime.h>
#include <hip/hip_bf16.h>
#include <math.h>

// ---------------------------------------------------------------------------
// KAN layer as one fused GEMM (M=8192, N=1024, K=8192, ~137 GFLOP):
//   A[n, i*8+k] = k<7 ? Bspline_k(clamp(x[n,i])) : silu(clamp(x[n,i]))   (f16)
//   Wp[o, i*8+k] = k<7 ? coeff[o,i,k] : w_base[o,i]                      (f16)
//   out[n,o] = sum_K A[n,K]*Wp[o,K] + bias[o]        (f32 accum via WMMA)
// Workspace requirement: 8192*8192*2 + 1024*8192*2 = ~151 MB in d_ws.
// GEMM: both operands staged in double-buffered LDS; async global->LDS
// copies (ASYNCcnt) overlap the WMMA stream.
// ---------------------------------------------------------------------------

typedef __attribute__((ext_vector_type(16))) _Float16 v16h;
typedef __attribute__((ext_vector_type(8)))  float    v8f;
typedef __attribute__((__vector_size__(4 * sizeof(int)))) int v4i;

#define NSAMP 8192
#define DIN   1024
#define DOUT  1024
#define KDIM  (DIN * 8)   // 8192

// ---------------- Pass 1: expand x -> 8 f16 features per element ----------
__device__ __forceinline__ void kan_features(float xin, _Float16 out8[8]) {
    const float t[11] = {-1.f,-1.f,-1.f,-1.f,-0.5f,0.f,0.5f,1.f,1.f,1.f,1.f};
    float xc = fminf(fmaxf(xin, -1.f), 1.f);
    float b[10];
#pragma unroll
    for (int j = 0; j < 10; ++j)
        b[j] = (xc >= t[j] && xc < t[j + 1]) ? 1.f : 0.f;
    if (xc == 1.f) b[9] += 1.f;                 // right-closed last interval
#pragma unroll
    for (int k = 1; k <= 3; ++k) {
#pragma unroll
        for (int j = 0; j < 10 - k; ++j) {
            float dl = fmaxf(t[j + k] - t[j], 1e-8f);       // constant-folded
            float cl = (xc - t[j]) / dl;
            float dr = fmaxf(t[j + k + 1] - t[j + 1], 1e-8f);
            float cr = (t[j + k + 1] - xc) / dr;
            b[j] = cl * b[j] + cr * b[j + 1];
        }
    }
#pragma unroll
    for (int k = 0; k < 7; ++k) out8[k] = (_Float16)b[k];
    float s = xc / (1.f + __expf(-xc));         // silu
    out8[7] = (_Float16)s;
}

__global__ __launch_bounds__(256) void expand_x_kernel(
        const float* __restrict__ x, _Float16* __restrict__ A) {
    size_t idx = (size_t)blockIdx.x * blockDim.x + threadIdx.x; // (n,i) flat
    float v = x[idx];
    _Float16 f[8];
    kan_features(v, f);
    union { _Float16 h[8]; uint4 u; } pk;
#pragma unroll
    for (int k = 0; k < 8; ++k) pk.h[k] = f[k];
    ((uint4*)A)[idx] = pk.u;   // A chunk (n*DIN + i) == idx, 16B coalesced
}

// ---------------- Pass 2: pack coeff + w_base -> f16 Wp --------------------
__global__ __launch_bounds__(256) void pack_w_kernel(
        const float* __restrict__ coeff, const float* __restrict__ w_base,
        _Float16* __restrict__ Wp) {
    size_t idx = (size_t)blockIdx.x * blockDim.x + threadIdx.x; // (o,i) flat
    const float* c = coeff + idx * 7;
    union { _Float16 h[8]; uint4 u; } pk;
#pragma unroll
    for (int k = 0; k < 7; ++k) pk.h[k] = (_Float16)c[k];
    pk.h[7] = (_Float16)w_base[idx];
    ((uint4*)Wp)[idx] = pk.u;  // Wp[o, i*8 .. i*8+7]
}

// ---------------- Pass 3: WMMA GEMM ----------------------------------------
// Block: 256 thr = 8 waves; block tile 128(M) x 128(N); wave tile 32x64.
// Both A and B tiles live in LDS (padded stride 72 halves -> conflict-free
// ds_load_b128 fragment reads), double-buffered; global->LDS copies go
// through the gfx1250 async path when available.

#define BM 128
#define BN 128
#define BK 64
#define LDT 72                 // padded LDS row stride in halves
#define TILE_H (BM * LDT)      // halves per tile buffer (9216 -> 18 KB)
#define KTILES (KDIM / BK)     // 128

#if __has_builtin(__builtin_amdgcn_global_load_async_to_lds_b128) && \
    __has_builtin(__builtin_amdgcn_s_wait_asynccnt)
#define USE_ASYNC_LDS 1
#else
#define USE_ASYNC_LDS 0
#endif

__device__ __forceinline__ void stage16(const _Float16* __restrict__ g,
                                        _Float16* l) {
#if USE_ASYNC_LDS
    // GLOBAL_LOAD_ASYNC_TO_LDS_B128: LDS[l] = MEM[g], tracked by ASYNCcnt.
    __builtin_amdgcn_global_load_async_to_lds_b128(
        (__attribute__((address_space(1))) v4i*)(g),
        (__attribute__((address_space(3))) v4i*)(l),
        0, 0);
#else
    *(uint4*)l = *(const uint4*)g;   // sync fallback: load + ds_store
#endif
}

union AFrag { uint4 u[2]; v16h v; };

__global__ __launch_bounds__(256) void kan_gemm_kernel(
        const _Float16* __restrict__ A, const _Float16* __restrict__ Wp,
        const float* __restrict__ bias, float* __restrict__ out) {
    __shared__ _Float16 sA[2 * TILE_H];   // 36 KB, ping-pong
    __shared__ _Float16 sB[2 * TILE_H];   // 36 KB, ping-pong

    const int tid  = threadIdx.x;
    const int lane = tid & 31;
    const int wave = tid >> 5;
    const int wm   = wave & 3;      // 4 waves along M -> 32 rows each
    const int wn   = wave >> 2;     // 2 waves along N -> 64 cols each
    const int l16  = lane & 15;
    const int lhi  = lane >> 4;     // half-wave selector (ISA frag layout)
    const int bm0  = blockIdx.y * BM;
    const int bn0  = blockIdx.x * BN;

    v8f acc[2][4];
#pragma unroll
    for (int mt = 0; mt < 2; ++mt)
#pragma unroll
        for (int nt = 0; nt < 4; ++nt)
            acc[mt][nt] = (v8f){0.f,0.f,0.f,0.f,0.f,0.f,0.f,0.f};

    // Cooperative stage of one 128x64-half tile of A and of B (8 async
    // instructions per wave -> ASYNCcnt += 8).
    auto stage_tiles = [&](int buf, int kk) {
#pragma unroll
        for (int c = 0; c < 4; ++c) {
            int chunk = tid + c * 256;          // 0..1023
            int row   = chunk >> 3;             // 0..127
            int cu    = chunk & 7;              // 16B chunk within row
            stage16(A  + (size_t)(bm0 + row) * KDIM + kk + cu * 8,
                    sA + buf * TILE_H + row * LDT + cu * 8);
            stage16(Wp + (size_t)(bn0 + row) * KDIM + kk + cu * 8,
                    sB + buf * TILE_H + row * LDT + cu * 8);
        }
    };

    stage_tiles(0, 0);                          // prologue fill

    for (int t = 0; t < KTILES; ++t) {
        const int buf = t & 1;
        if (t + 1 < KTILES) {
            stage_tiles(buf ^ 1, (t + 1) * BK); // overlaps this tile's WMMAs
#if USE_ASYNC_LDS
            __builtin_amdgcn_s_wait_asynccnt(8); // tile t's 8 copies done
#endif
        } else {
#if USE_ASYNC_LDS
            __builtin_amdgcn_s_wait_asynccnt(0);
#endif
        }
        __syncthreads();                        // publish tile t block-wide

        const _Float16* tA = sA + buf * TILE_H;
        const _Float16* tB = sB + buf * TILE_H;
#pragma unroll
        for (int k0 = 0; k0 < BK; k0 += 32) {
            // A fragments (16x32 f16): lane holds K=kb..kb+7 and kb+16..kb+23
            AFrag a[2];
#pragma unroll
            for (int mt = 0; mt < 2; ++mt) {
                int r  = wm * 32 + mt * 16 + l16;
                int kb = k0 + lhi * 8;
                a[mt].u[0] = *(const uint4*)(tA + r * LDT + kb);
                a[mt].u[1] = *(const uint4*)(tA + r * LDT + kb + 16);
            }
#pragma unroll
            for (int nt = 0; nt < 4; ++nt) {
                // B fragment (32x16): lane = column, K=kb..kb+7 and kb+16..kb+23
                int c  = wn * 64 + nt * 16 + l16;
                int kb = k0 + lhi * 8;
                AFrag b;
                b.u[0] = *(const uint4*)(tB + c * LDT + kb);
                b.u[1] = *(const uint4*)(tB + c * LDT + kb + 16);
#pragma unroll
                for (int mt = 0; mt < 2; ++mt)
                    acc[mt][nt] = __builtin_amdgcn_wmma_f32_16x16x32_f16(
                        false, a[mt].v, false, b.v,
                        (short)0, acc[mt][nt], false, false);
            }
        }
        __syncthreads();                        // done reading buf before reuse
    }

    // ---- epilogue: C layout = lane -> column, VGPR v -> row (+8 hi half) --
#pragma unroll
    for (int nt = 0; nt < 4; ++nt) {
        int col  = bn0 + wn * 64 + nt * 16 + l16;
        float bs = bias[col];
#pragma unroll
        for (int mt = 0; mt < 2; ++mt) {
            int rbase = bm0 + wm * 32 + mt * 16 + lhi * 8;
#pragma unroll
            for (int v = 0; v < 8; ++v)
                out[(size_t)(rbase + v) * DOUT + col] = acc[mt][nt][v] + bs;
        }
    }
}

// ---------------------------------------------------------------------------
extern "C" void kernel_launch(void* const* d_in, const int* in_sizes, int n_in,
                              void* d_out, int out_size, void* d_ws, size_t ws_size,
                              hipStream_t stream) {
    (void)in_sizes; (void)n_in; (void)out_size; (void)ws_size;
    const float* x      = (const float*)d_in[0];   // (8192, 1024)
    const float* coeff  = (const float*)d_in[1];   // (1024, 1024, 7)
    const float* w_base = (const float*)d_in[2];   // (1024, 1024)
    const float* bias   = (const float*)d_in[3];   // (1024,)
    float* out = (float*)d_out;                    // (8192, 1024)

    _Float16* A  = (_Float16*)d_ws;                                   // 134 MB
    _Float16* Wp = (_Float16*)((char*)d_ws + (size_t)NSAMP * KDIM * 2); // +16 MB

    expand_x_kernel<<<(NSAMP * DIN) / 256, 256, 0, stream>>>(x, A);
    pack_w_kernel<<<(DOUT * DIN) / 256, 256, 0, stream>>>(coeff, w_base, Wp);

    dim3 grid(DOUT / BN, NSAMP / BM);   // (8, 64)
    kan_gemm_kernel<<<grid, 256, 0, stream>>>(A, Wp, bias, out);
}
